// SetAbstractionMrgSeq_44659069944097
// MI455X (gfx1250) — matrix operand order repaired and checked by reading, not compiled, not run on hardware
//
#include <hip/hip_runtime.h>

typedef _Float16 h16;
typedef __attribute__((ext_vector_type(16))) _Float16 v16h;
typedef __attribute__((ext_vector_type(8)))  _Float16 v8h;
typedef __attribute__((ext_vector_type(8)))  float    v8f;

// ---------------------------------------------------------------------------
// WMMA tile core: acc += A(16x32, LDS) x B(32x16, pre-packed global)
// Weights pre-packed in fragment order:
//   Wp[((ks*CT + ct)*32 + lane)*16 + e] = W[ks*32 + (lane>>4)*16 + e][ct*16 + (lane&15)]
// so each lane's B fragment is one contiguous 32B (v16h) load.
// ---------------------------------------------------------------------------
__device__ __forceinline__ v8f wmma_tile_acc(
    const h16* __restrict__ lds_in, int ldin, int rt, int ct, int CT, int KS,
    const h16* __restrict__ Wp, int lane)
{
    const int lrow = lane & 15, lg = lane >> 4;
    v8f acc = {};
    for (int ks = 0; ks < KS; ++ks) {
        const int arow = rt * 16 + lrow;
        const int kb   = ks * 32 + lg * 16;
        v8h alo = *(const v8h*)(lds_in + (size_t)arow * ldin + kb);
        v8h ahi = *(const v8h*)(lds_in + (size_t)arow * ldin + kb + 8);
        v16h a;
#pragma unroll
        for (int i = 0; i < 8; ++i) { a[i] = alo[i]; a[i + 8] = ahi[i]; }
        v16h bf = *(const v16h*)(Wp + ((size_t)(ks * CT + ct) * 32 + lane) * 16);
        acc = __builtin_amdgcn_wmma_f32_16x16x32_f16(
            false, a, false, bf, (short)0, acc, false, false);
    }
    return acc;
}

// Intermediate layer: out = relu(in @ W + bias) -> LDS f16
__device__ __forceinline__ void wmma_layer(
    const h16* __restrict__ lds_in, int ldin,
    h16* __restrict__ lds_out, int ldout,
    int rows, int Cin, int Cout,
    const h16* __restrict__ Wp, const float* __restrict__ bias)
{
    const int wv = threadIdx.x >> 5, lane = threadIdx.x & 31, nw = blockDim.x >> 5;
    const int RT = rows >> 4, CT = Cout >> 4, KS = Cin >> 5;
    const int lrow = lane & 15, lg = lane >> 4;
    for (int tile = wv; tile < RT * CT; tile += nw) {
        const int rt = tile / CT, ct = tile % CT;
        const int col = ct * 16 + lrow;
        v8f acc = wmma_tile_acc(lds_in, ldin, rt, ct, CT, KS, Wp, lane);
        const float bv = bias[col];
#pragma unroll
        for (int r = 0; r < 8; ++r) {
            float v = acc[r] + bv;
            v = v > 0.f ? v : 0.f;                 // relu (plain_last=False)
            const int orow = rt * 16 + lg * 8 + r; // D layout: VGPR r rows
            lds_out[(size_t)orow * ldout + col] = (h16)v;
        }
    }
}

// Final layer fused with masked max-pool over rows:
//   res[c] = max_{k < kc} relu( (in @ W + bias)[k][c] )
// Since relu >= 0 and kc >= 1, identity 0 is a safe mask/seed.
// lds_pool: RT*Cout floats scratch. If atomic_out: atomicMax(uint bits) into
// gout (valid for non-negative floats); else plain store.
__device__ __forceinline__ void wmma_layer_pool(
    const h16* __restrict__ lds_in, int ldin,
    int rows, int Cin, int Cout,
    const h16* __restrict__ Wp, const float* __restrict__ bias,
    int kc, float* __restrict__ lds_pool,
    float* __restrict__ gout, bool atomic_out)
{
    const int wv = threadIdx.x >> 5, lane = threadIdx.x & 31, nw = blockDim.x >> 5;
    const int RT = rows >> 4, CT = Cout >> 4, KS = Cin >> 5;
    const int lrow = lane & 15, lg = lane >> 4;
    for (int tile = wv; tile < RT * CT; tile += nw) {
        const int rt = tile / CT, ct = tile % CT;
        const int col = ct * 16 + lrow;
        v8f acc = wmma_tile_acc(lds_in, ldin, rt, ct, CT, KS, Wp, lane);
        const float bv = bias[col];
        float pmax = 0.f;
#pragma unroll
        for (int r = 0; r < 8; ++r) {
            float v = acc[r] + bv;
            v = v > 0.f ? v : 0.f;
            const int orow = rt * 16 + lg * 8 + r;
            if (orow < kc && v > pmax) pmax = v;   // mask padded rows -> 0
        }
        // lanes L and L+16 hold the two row-halves of the same column
        const float other = __shfl_xor(pmax, 16, 32);
        pmax = other > pmax ? other : pmax;
        if (lg == 0) lds_pool[rt * Cout + col] = pmax;
    }
    __syncthreads();
    for (int c = threadIdx.x; c < Cout; c += blockDim.x) {
        float mx = 0.f;
        for (int rt = 0; rt < RT; ++rt) {
            const float v = lds_pool[rt * Cout + c];
            mx = v > mx ? v : mx;
        }
        if (atomic_out) atomicMax((unsigned*)&gout[c], __float_as_uint(mx));
        else            gout[c] = mx;
    }
    __syncthreads();
}

// ---------------------------------------------------------------------------
// Utility kernels
// ---------------------------------------------------------------------------
__global__ void init_out_kernel(float* __restrict__ out, int n) {
    int i = blockIdx.x * blockDim.x + threadIdx.x;
    if (i < n) out[i] = 0.f;
}

// f32 weights -> f16 packed in WMMA B-fragment order, Cin padded to Kpad.
__global__ void convert_w_kernel(const float* __restrict__ src, h16* __restrict__ dst,
                                 int Cin, int Cout, int Kpad) {
    const int CT = Cout >> 4;
    const int total = Kpad * Cout;
    for (int i = blockIdx.x * blockDim.x + threadIdx.x; i < total;
         i += gridDim.x * blockDim.x) {
        const int e    = i & 15;
        const int lane = (i >> 4) & 31;
        const int tile = i >> 9;            // ks*CT + ct
        const int ct = tile % CT, ks = tile / CT;
        const int srow = ks * 32 + (lane >> 4) * 16 + e;
        const int scol = ct * 16 + (lane & 15);
        dst[i] = (srow < Cin) ? (h16)src[srow * Cout + scol] : (h16)0.f;
    }
}

// Farthest-point sampling: one block per batch; sequential M steps,
// parallel dmin update + argmax reduction (ties -> smallest index).
__global__ __launch_bounds__(1024) void fps_kernel(
    const float* __restrict__ p, int N, int M, int* __restrict__ sidx)
{
    __shared__ float rd[1024];
    __shared__ int   ri[1024];
    __shared__ float lastpt[2];
    const int b = blockIdx.x;
    const float* pb = p + (size_t)b * N * 2;
    const int t = threadIdx.x;

    float dmin[8];
#pragma unroll
    for (int i = 0; i < 8; ++i) dmin[i] = 3.4e38f;
    int last = 0;

    for (int m = 0; m < M; ++m) {
        if (t == 0) {
            sidx[b * M + m] = last;
            lastpt[0] = pb[last * 2 + 0];
            lastpt[1] = pb[last * 2 + 1];
        }
        __syncthreads();
        const float lx = lastpt[0], ly = lastpt[1];
        float bestd = -1.f; int besti = 0;
#pragma unroll
        for (int i = 0; i < 8; ++i) {
            const int n = t + i * 1024;
            if (n < N) {
                const float dx = pb[n * 2] - lx, dy = pb[n * 2 + 1] - ly;
                const float d = dx * dx + dy * dy;
                if (d < dmin[i]) dmin[i] = d;
                if (dmin[i] > bestd) { bestd = dmin[i]; besti = n; } // ascending n
            }
        }
        rd[t] = bestd; ri[t] = besti;
        __syncthreads();
        for (int s = 512; s > 0; s >>= 1) {
            if (t < s) {
                if (rd[t + s] > rd[t] ||
                    (rd[t + s] == rd[t] && ri[t + s] < ri[t])) {
                    rd[t] = rd[t + s]; ri[t] = ri[t + s];
                }
            }
            __syncthreads();
        }
        last = ri[0];
        __syncthreads();
    }
}

__global__ void gather_q_kernel(const float* __restrict__ p, const int* __restrict__ sidx,
                                float* __restrict__ q, int N, int M, int BM) {
    int i = blockIdx.x * blockDim.x + threadIdx.x;
    if (i < BM) {
        const int b = i / M;
        const int s = sidx[i];
        q[i * 2 + 0] = p[((size_t)b * N + s) * 2 + 0];
        q[i * 2 + 1] = p[((size_t)b * N + s) * 2 + 1];
    }
}

// Ball grouping: one wave per query; first-64-by-index in-radius neighbors
// via ballot + prefix popcount (wave32).
__global__ __launch_bounds__(128) void ball_group_kernel(
    const float* __restrict__ q, const float* __restrict__ p,
    int M, int N, float r2, int BM,
    int* __restrict__ nidx, int* __restrict__ cnt)
{
    const int wv = threadIdx.x >> 5, lane = threadIdx.x & 31;
    const int gq = blockIdx.x * (blockDim.x >> 5) + wv;
    if (gq >= BM) return;
    const float qx = q[gq * 2], qy = q[gq * 2 + 1];
    const float* pb = p + (size_t)(gq / M) * N * 2;
    int* out = nidx + (size_t)gq * 64;
    int c = 0;
    for (int j0 = 0; j0 < N && c < 64; j0 += 32) {
        const int j = j0 + lane;
        bool in = false;
        if (j < N) {
            const float dx = pb[j * 2] - qx, dy = pb[j * 2 + 1] - qy;
            in = (dx * dx + dy * dy) <= r2;
        }
        const unsigned mlo = (unsigned)__ballot(in);
        const int pos = __popc(mlo & ((1u << lane) - 1u));
        if (in && (c + pos) < 64) out[c + pos] = j;
        c += __popc(mlo);
    }
    if (c > 64) c = 64;
    for (int s = c + lane; s < 64; s += 32) out[s] = N - 1; // pad = min(idx,N-1)
    if (lane == 0) cnt[gq] = c;
}

// ---------------------------------------------------------------------------
// Level-1 fused kernel: b1s1 & b2 share FPS/grouping; both MLP chains in LDS;
// final layers fused with masked max-pool. One workgroup per query, 64 rows.
// LDS: 4 + 8 + 16 + 4 = 32 KB.
// ---------------------------------------------------------------------------
__global__ __launch_bounds__(128) void level1_kernel(
    const float* __restrict__ x, const float* __restrict__ pos,
    const float* __restrict__ q1, const int* __restrict__ nidx,
    const int* __restrict__ cnt,
    const h16* __restrict__ w11, const float* __restrict__ b11,   // 32x64
    const h16* __restrict__ w12, const float* __restrict__ b12,   // 64x128
    const h16* __restrict__ w21, const float* __restrict__ b21,   // 32x64
    const h16* __restrict__ w22, const float* __restrict__ b22,   // 64x128
    const h16* __restrict__ w23, const float* __restrict__ b23,   // 128x256
    float* __restrict__ x1a, float* __restrict__ x2,
    int N, int M)
{
    __shared__ __align__(16) h16 bufIn[64 * 32];
    __shared__ __align__(16) h16 bufA[64 * 64];
    __shared__ __align__(16) h16 bufB[64 * 128];
    __shared__ __align__(16) float pool[4 * 256];

    const int gq = blockIdx.x;            // b*M + m
    const int b  = gq / M;
    const int* nb = nidx + (size_t)gq * 64;
    const float qx = q1[gq * 2], qy = q1[gq * 2 + 1];
    const float* xb = x + (size_t)b * N * 3;
    const float* pb = pos + (size_t)b * N * 2;

    for (int i = threadIdx.x; i < 64 * 32; i += blockDim.x) {
        const int k = i >> 5, c = i & 31;
        const int n = nb[k];
        float v;
        if (c < 3)       v = xb[n * 3 + c];
        else if (c < 5)  v = pb[n * 2 + (c - 3)] - (c == 3 ? qx : qy);
        else             v = 0.f;
        bufIn[i] = (h16)v;
    }
    __syncthreads();
    const int kc = cnt[gq];

    // branch 1 stage 1: 5->64->128, pool -> x1a
    wmma_layer(bufIn, 32, bufA, 64, 64, 32, 64, w11, b11);
    __syncthreads();
    wmma_layer_pool(bufA, 64, 64, 64, 128, w12, b12, kc, pool,
                    x1a + (size_t)gq * 128, false);
    // branch 2: 5->64->128->256, pool -> x2 (reuses gathered input)
    wmma_layer(bufIn, 32, bufA, 64, 64, 32, 64, w21, b21);
    __syncthreads();
    wmma_layer(bufA, 64, bufB, 128, 64, 64, 128, w22, b22);
    __syncthreads();
    wmma_layer_pool(bufB, 128, 64, 128, 256, w23, b23, kc, pool,
                    x2 + (size_t)gq * 256, false);
}

// Level-2 (b1s2): gather 128+2 feats (pad->160), 130->256 fused with pool.
// LDS: 20 + 4 = 24 KB.
__global__ __launch_bounds__(128) void level2_kernel(
    const float* __restrict__ x1a, const float* __restrict__ p1,
    const float* __restrict__ q2, const int* __restrict__ nidx2,
    const int* __restrict__ cnt2,
    const h16* __restrict__ w, const float* __restrict__ bias,   // 160x256 packed
    float* __restrict__ x1f, int N, int M)
{
    __shared__ __align__(16) h16 bufIn[64 * 160];
    __shared__ __align__(16) float pool[4 * 256];
    const int gq = blockIdx.x;
    const int b  = gq / M;
    const int* nb = nidx2 + (size_t)gq * 64;
    const float qx = q2[gq * 2], qy = q2[gq * 2 + 1];
    const float* xb = x1a + (size_t)b * N * 128;
    const float* pb = p1 + (size_t)b * N * 2;

    for (int i = threadIdx.x; i < 64 * 160; i += blockDim.x) {
        const int k = i / 160, c = i % 160;
        const int n = nb[k];
        float v;
        if (c < 128)      v = xb[(size_t)n * 128 + c];
        else if (c < 130) v = pb[n * 2 + (c - 128)] - (c == 128 ? qx : qy);
        else              v = 0.f;
        bufIn[i] = (h16)v;
    }
    __syncthreads();
    wmma_layer_pool(bufIn, 160, 64, 160, 256, w, bias, cnt2[gq], pool,
                    x1f + (size_t)gq * 256, false);
}

// Branch 3: per-point MLP 5->128->256->512, global max per batch (atomic).
// 32 rows per workgroup. LDS: 18 + 8 + 4 = 30 KB.
__global__ __launch_bounds__(128) void branch3_kernel(
    const float* __restrict__ x, const float* __restrict__ pos,
    const h16* __restrict__ w1, const float* __restrict__ b1,
    const h16* __restrict__ w2, const float* __restrict__ b2,
    const h16* __restrict__ w3, const float* __restrict__ b3,
    float* __restrict__ outmax, int N)
{
    __shared__ __align__(16) h16 bufA[32 * 288];   // input(32x32) then L2 out(32x256)
    __shared__ __align__(16) h16 bufB[32 * 128];   // L1 out
    __shared__ __align__(16) float pool[2 * 512];
    const int tiles_per_b = N / 32;
    const int b  = blockIdx.x / tiles_per_b;
    const int r0 = (blockIdx.x % tiles_per_b) * 32;
    const float* xb = x + (size_t)b * N * 3;
    const float* pb = pos + (size_t)b * N * 2;

    for (int i = threadIdx.x; i < 32 * 32; i += blockDim.x) {
        const int k = i >> 5, c = i & 31;
        const int n = r0 + k;
        const float v = c < 3 ? xb[n * 3 + c] : (c < 5 ? pb[n * 2 + (c - 3)] : 0.f);
        bufA[k * 32 + c] = (h16)v;
    }
    __syncthreads();
    wmma_layer(bufA, 32, bufB, 128, 32, 32, 128, w1, b1);
    __syncthreads();
    wmma_layer(bufB, 128, bufA, 256, 32, 128, 256, w2, b2);
    __syncthreads();
    wmma_layer_pool(bufA, 256, 32, 256, 512, w3, b3, 32, pool,
                    outmax + b * 1024, true);
}

// Branch 4: rows = [x1f|q2](512) ++ [x2|q1](4096) -> 258 feats (pad 288),
// 258->512 fused with pool, atomic max per batch. LDS: 18 + 4 = 22 KB.
__global__ __launch_bounds__(128) void branch4_kernel(
    const float* __restrict__ x1f, const float* __restrict__ q2,
    const float* __restrict__ x2,  const float* __restrict__ q1,
    const h16* __restrict__ w, const float* __restrict__ bias,   // 288x512 packed
    float* __restrict__ outmax)
{
    __shared__ __align__(16) h16 bufIn[32 * 288];
    __shared__ __align__(16) float pool[2 * 512];
    const int tiles_per_b = 4608 / 32;   // 144
    const int b  = blockIdx.x / tiles_per_b;
    const int j0 = (blockIdx.x % tiles_per_b) * 32;

    for (int i = threadIdx.x; i < 32 * 288; i += blockDim.x) {
        const int k = i / 288, c = i % 288;
        const int j = j0 + k;
        float v = 0.f;
        if (j < 512) {
            if (c < 256)      v = x1f[((size_t)b * 512 + j) * 256 + c];
            else if (c < 258) v = q2[((size_t)b * 512 + j) * 2 + (c - 256)];
        } else {
            const int jj = j - 512;
            if (c < 256)      v = x2[((size_t)b * 4096 + jj) * 256 + c];
            else if (c < 258) v = q1[((size_t)b * 4096 + jj) * 2 + (c - 256)];
        }
        bufIn[i] = (h16)v;
    }
    __syncthreads();
    wmma_layer_pool(bufIn, 288, 32, 288, 512, w, bias, 32, pool,
                    outmax + b * 1024 + 512, true);
}

// ---------------------------------------------------------------------------
// Host-side orchestration
// ---------------------------------------------------------------------------
extern "C" void kernel_launch(void* const* d_in, const int* in_sizes, int n_in,
                              void* d_out, int out_size, void* d_ws, size_t ws_size,
                              hipStream_t stream) {
    (void)in_sizes; (void)n_in; (void)ws_size;
    const int B = 2, N = 8192, M1 = 4096, M2 = 512;

    const float* x   = (const float*)d_in[0];
    const float* pos = (const float*)d_in[1];
    // params pytree order: b1s1[(w,b),(w,b)], b1s2[(w,b)], b2[3x(w,b)],
    //                      b3[3x(w,b)], b4[(w,b)]
    const float* b1s1_w1 = (const float*)d_in[2];
    const float* b1s1_b1 = (const float*)d_in[3];
    const float* b1s1_w2 = (const float*)d_in[4];
    const float* b1s1_b2 = (const float*)d_in[5];
    const float* b1s2_w  = (const float*)d_in[6];
    const float* b1s2_b  = (const float*)d_in[7];
    const float* b2_w1 = (const float*)d_in[8];
    const float* b2_b1 = (const float*)d_in[9];
    const float* b2_w2 = (const float*)d_in[10];
    const float* b2_b2 = (const float*)d_in[11];
    const float* b2_w3 = (const float*)d_in[12];
    const float* b2_b3 = (const float*)d_in[13];
    const float* b3_w1 = (const float*)d_in[14];
    const float* b3_b1 = (const float*)d_in[15];
    const float* b3_w2 = (const float*)d_in[16];
    const float* b3_b2 = (const float*)d_in[17];
    const float* b3_w3 = (const float*)d_in[18];
    const float* b3_b3 = (const float*)d_in[19];
    const float* b4_w  = (const float*)d_in[20];
    const float* b4_b  = (const float*)d_in[21];

    float* out = (float*)d_out;

    // bump allocator on workspace
    char* wsp = (char*)d_ws;
    auto alloc = [&](size_t bytes) -> void* {
        void* p = wsp;
        wsp += (bytes + 255) & ~(size_t)255;
        return p;
    };
    int*   sidx1 = (int*)alloc((size_t)B * M1 * 4);
    float* q1    = (float*)alloc((size_t)B * M1 * 2 * 4);
    int*   nidx1 = (int*)alloc((size_t)B * M1 * 64 * 4);
    int*   cnt1  = (int*)alloc((size_t)B * M1 * 4);
    float* x1a   = (float*)alloc((size_t)B * M1 * 128 * 4);
    float* x2    = (float*)alloc((size_t)B * M1 * 256 * 4);
    int*   sidx2 = (int*)alloc((size_t)B * M2 * 4);
    float* q2    = (float*)alloc((size_t)B * M2 * 2 * 4);
    int*   nidx2 = (int*)alloc((size_t)B * M2 * 64 * 4);
    int*   cnt2  = (int*)alloc((size_t)B * M2 * 4);
    float* x1f   = (float*)alloc((size_t)B * M2 * 256 * 4);
    h16* fw_b1s1_1 = (h16*)alloc((size_t)32 * 64 * 2);
    h16* fw_b1s1_2 = (h16*)alloc((size_t)64 * 128 * 2);
    h16* fw_b1s2   = (h16*)alloc((size_t)160 * 256 * 2);
    h16* fw_b2_1   = (h16*)alloc((size_t)32 * 64 * 2);
    h16* fw_b2_2   = (h16*)alloc((size_t)64 * 128 * 2);
    h16* fw_b2_3   = (h16*)alloc((size_t)128 * 256 * 2);
    h16* fw_b3_1   = (h16*)alloc((size_t)32 * 128 * 2);
    h16* fw_b3_2   = (h16*)alloc((size_t)128 * 256 * 2);
    h16* fw_b3_3   = (h16*)alloc((size_t)256 * 512 * 2);
    h16* fw_b4     = (h16*)alloc((size_t)288 * 512 * 2);

    // outputs zeroed for atomic-max accumulation
    init_out_kernel<<<(out_size + 255) / 256, 256, 0, stream>>>(out, out_size);

    // weight conversion: f16 + Cin padding + WMMA fragment packing
    convert_w_kernel<<<16, 256, 0, stream>>>(b1s1_w1, fw_b1s1_1, 5, 64, 32);
    convert_w_kernel<<<32, 256, 0, stream>>>(b1s1_w2, fw_b1s1_2, 64, 128, 64);
    convert_w_kernel<<<64, 256, 0, stream>>>(b1s2_w,  fw_b1s2, 130, 256, 160);
    convert_w_kernel<<<16, 256, 0, stream>>>(b2_w1, fw_b2_1, 5, 64, 32);
    convert_w_kernel<<<32, 256, 0, stream>>>(b2_w2, fw_b2_2, 64, 128, 64);
    convert_w_kernel<<<64, 256, 0, stream>>>(b2_w3, fw_b2_3, 128, 256, 128);
    convert_w_kernel<<<16, 256, 0, stream>>>(b3_w1, fw_b3_1, 5, 128, 32);
    convert_w_kernel<<<64, 256, 0, stream>>>(b3_w2, fw_b3_2, 128, 256, 128);
    convert_w_kernel<<<128, 256, 0, stream>>>(b3_w3, fw_b3_3, 256, 512, 256);
    convert_w_kernel<<<128, 256, 0, stream>>>(b4_w,  fw_b4, 258, 512, 288);

    // level 1 sampling/grouping (shared by b1s1 and b2)
    fps_kernel<<<B, 1024, 0, stream>>>(pos, N, M1, sidx1);
    gather_q_kernel<<<(B * M1 + 255) / 256, 256, 0, stream>>>(pos, sidx1, q1, N, M1, B * M1);
    ball_group_kernel<<<(B * M1 + 3) / 4, 128, 0, stream>>>(q1, pos, M1, N, 0.36f,
                                                            B * M1, nidx1, cnt1);
    // fused grouped MLPs + max-pool for b1s1 and b2
    level1_kernel<<<B * M1, 128, 0, stream>>>(
        x, pos, q1, nidx1, cnt1,
        fw_b1s1_1, b1s1_b1, fw_b1s1_2, b1s1_b2,
        fw_b2_1, b2_b1, fw_b2_2, b2_b2, fw_b2_3, b2_b3,
        x1a, x2, N, M1);

    // level 2 (b1s2) on (x1a, p1=q1)
    fps_kernel<<<B, 1024, 0, stream>>>(q1, M1, M2, sidx2);
    gather_q_kernel<<<(B * M2 + 255) / 256, 256, 0, stream>>>(q1, sidx2, q2, M1, M2, B * M2);
    ball_group_kernel<<<(B * M2 + 3) / 4, 128, 0, stream>>>(q2, q1, M2, M1, 0.64f,
                                                            B * M2, nidx2, cnt2);
    level2_kernel<<<B * M2, 128, 0, stream>>>(x1a, q1, q2, nidx2, cnt2,
                                              fw_b1s2, b1s2_b, x1f, M1, M2);

    // global branches (atomic-max into d_out)
    branch3_kernel<<<B * (N / 32), 128, 0, stream>>>(
        x, pos, fw_b3_1, b3_b1, fw_b3_2, b3_b2, fw_b3_3, b3_b3, out, N);
    branch4_kernel<<<B * (4608 / 32), 128, 0, stream>>>(
        x1f, q2, x2, q1, fw_b4, b4_b, out);
}
